// BiMambaWrapper_57423712747976
// MI455X (gfx1250) — compile-verified, matrix operand order checked by reading, not run on hardware
//
#include <hip/hip_runtime.h>

typedef __attribute__((ext_vector_type(16))) __bf16 v16bf;
typedef __attribute__((ext_vector_type(8)))  float  v8f;

#define BATCH   2
#define SEQLEN  1024
#define DMODEL  768
#define DINNER  1536
#define DSTATE  16
#define DTRANK  48
#define NTOK    (BATCH * SEQLEN)      // 2048 rows for all GEMMs

// ---------------------------------------------------------------------------
// helpers
// ---------------------------------------------------------------------------
__device__ __forceinline__ v16bf pack_bf16(float4 a0, float4 a1, float4 a2, float4 a3) {
    v16bf v;
    v[0]  = (__bf16)a0.x; v[1]  = (__bf16)a0.y; v[2]  = (__bf16)a0.z; v[3]  = (__bf16)a0.w;
    v[4]  = (__bf16)a1.x; v[5]  = (__bf16)a1.y; v[6]  = (__bf16)a1.z; v[7]  = (__bf16)a1.w;
    v[8]  = (__bf16)a2.x; v[9]  = (__bf16)a2.y; v[10] = (__bf16)a2.z; v[11] = (__bf16)a2.w;
    v[12] = (__bf16)a3.x; v[13] = (__bf16)a3.y; v[14] = (__bf16)a3.z; v[15] = (__bf16)a3.w;
    return v;
}

// Unguarded fragment load: lanes 0-15 take K kA..kA+7 / kB..kB+7 per the
// gfx1250 16-bit A/B VGPR layout (kA = kb + kh*8, kB = kb + 16 + kh*8).
__device__ __forceinline__ v16bf load_frag(const float* row, int kA, int kB) {
    float4 a0 = *(const float4*)(row + kA);
    float4 a1 = *(const float4*)(row + kA + 4);
    float4 a2 = *(const float4*)(row + kB);
    float4 a3 = *(const float4*)(row + kB + 4);
    return pack_bf16(a0, a1, a2, a3);
}

// Masked tail fragment load: clamp the address (always in-bounds, K%8==0)
// and zero OOB halves with multiplies -> pure VALU select, no EXEC branches.
__device__ __forceinline__ v16bf load_frag_tail(const float* row, int kA, int kB, int K) {
    int   kAc = (kA + 8 <= K) ? kA : (K - 8);
    int   kBc = (kB + 8 <= K) ? kB : (K - 8);
    float mA  = (kA + 8 <= K) ? 1.f : 0.f;
    float mB  = (kB + 8 <= K) ? 1.f : 0.f;
    float4 a0 = *(const float4*)(row + kAc);
    float4 a1 = *(const float4*)(row + kAc + 4);
    float4 a2 = *(const float4*)(row + kBc);
    float4 a3 = *(const float4*)(row + kBc + 4);
    a0.x *= mA; a0.y *= mA; a0.z *= mA; a0.w *= mA;
    a1.x *= mA; a1.y *= mA; a1.z *= mA; a1.w *= mA;
    a2.x *= mB; a2.y *= mB; a2.z *= mB; a2.w *= mB;
    a3.x *= mB; a3.y *= mB; a3.z *= mB; a3.w *= mB;
    return pack_bf16(a0, a1, a2, a3);
}

__device__ __forceinline__ float silu_f(float x) { return x / (1.f + __expf(-x)); }

// ---------------------------------------------------------------------------
// WMMA GEMM:  C[M,N] = A[M,K] * B[N,K]^T   (row-major fp32 in/out, bf16 MACs,
// fp32 accumulate).  One wave32 computes a 16x64 strip: 4 independent
// accumulators share one A fragment.  sched_barriers keep the 4 WMMAs
// back-to-back so the WMMA->VALU WAR hazard (4 coexec NOPs for bf16) is paid
// once per K-iteration instead of once per WMMA.
// mode 0: plain store; mode 1: softplus(acc + bias[n]) (dt projection)
// Requires M%16==0, N%16==0, K%8==0.
// ---------------------------------------------------------------------------
__global__ __launch_bounds__(32)
void wmma_gemm_bf16(const float* __restrict__ A, const float* __restrict__ B,
                    const float* __restrict__ bias, float* __restrict__ C,
                    int M, int N, int K, int lda, int ldb, int ldc, int mode)
{
    const int m0 = blockIdx.x * 16;
    const int n0 = blockIdx.y * 64;
    if (m0 >= M || n0 >= N) return;                  // wave-uniform

    const int lane = threadIdx.x;                    // 0..31
    const int row  = lane & 15;
    const int kh   = lane >> 4;                      // K-half select

    const float* Arow = A + (size_t)(m0 + row) * lda;
    // Clamp OOB N-subtiles to the last valid tile: loads always legal,
    // results discarded at the store.  N%16==0 for all call sites.
    const float* Brow[4];
#pragma unroll
    for (int s = 0; s < 4; ++s) {
        int ns = n0 + s * 16;
        if (ns > N - 16) ns = N - 16;
        Brow[s] = B + (size_t)(ns + row) * ldb;
    }

    v8f acc0 = {}, acc1 = {}, acc2 = {}, acc3 = {};

    const int Kmain = K & ~31;
    int kb = 0;
    for (; kb < Kmain; kb += 32) {                   // unguarded hot loop
        const int kA = kb + kh * 8;
        const int kB = kb + 16 + kh * 8;
        v16bf av  = load_frag(Arow,    kA, kB);
        v16bf bv0 = load_frag(Brow[0], kA, kB);
        v16bf bv1 = load_frag(Brow[1], kA, kB);
        v16bf bv2 = load_frag(Brow[2], kA, kB);
        v16bf bv3 = load_frag(Brow[3], kA, kB);
        __builtin_amdgcn_sched_barrier(0);           // loads+cvts stay above
        acc0 = __builtin_amdgcn_wmma_f32_16x16x32_bf16(false, av, false, bv0, (short)0, acc0, false, false);
        acc1 = __builtin_amdgcn_wmma_f32_16x16x32_bf16(false, av, false, bv1, (short)0, acc1, false, false);
        acc2 = __builtin_amdgcn_wmma_f32_16x16x32_bf16(false, av, false, bv2, (short)0, acc2, false, false);
        acc3 = __builtin_amdgcn_wmma_f32_16x16x32_bf16(false, av, false, bv3, (short)0, acc3, false, false);
        __builtin_amdgcn_sched_barrier(0);           // WMMAs stay contiguous
    }
    if (kb < K) {                                    // masked tail (only K=48)
        const int kA = kb + kh * 8;
        const int kB = kb + 16 + kh * 8;
        v16bf av  = load_frag_tail(Arow,    kA, kB, K);
        v16bf bv0 = load_frag_tail(Brow[0], kA, kB, K);
        v16bf bv1 = load_frag_tail(Brow[1], kA, kB, K);
        v16bf bv2 = load_frag_tail(Brow[2], kA, kB, K);
        v16bf bv3 = load_frag_tail(Brow[3], kA, kB, K);
        __builtin_amdgcn_sched_barrier(0);
        acc0 = __builtin_amdgcn_wmma_f32_16x16x32_bf16(false, av, false, bv0, (short)0, acc0, false, false);
        acc1 = __builtin_amdgcn_wmma_f32_16x16x32_bf16(false, av, false, bv1, (short)0, acc1, false, false);
        acc2 = __builtin_amdgcn_wmma_f32_16x16x32_bf16(false, av, false, bv2, (short)0, acc2, false, false);
        acc3 = __builtin_amdgcn_wmma_f32_16x16x32_bf16(false, av, false, bv3, (short)0, acc3, false, false);
        __builtin_amdgcn_sched_barrier(0);
    }

    // C/D layout: VGPR r -> M = r + 8*(lane>=16), N = lane&15
    v8f accs[4] = {acc0, acc1, acc2, acc3};
#pragma unroll
    for (int s = 0; s < 4; ++s) {
        const int ns = n0 + s * 16;
        if (ns >= N) break;                          // wave-uniform
        const int n = ns + row;
        float bn = (mode == 1) ? bias[n] : 0.f;
#pragma unroll
        for (int r = 0; r < 8; ++r) {
            int   m = m0 + r + kh * 8;
            float v = accs[s][r];
            if (mode == 1) {
                float x = v + bn;
                v = (x > 20.f) ? x : log1pf(__expf(x));   // softplus
            }
            C[(size_t)m * ldc + n] = v;
        }
    }
}

// ---------------------------------------------------------------------------
// Depthwise causal conv1d (D_CONV=4) + bias + SiLU, both directions.
// xz rows are [x(1536) | z(1536)].  xt_r is produced in *reversed* sequence
// coordinates (t == position in the flipped sequence) so the scan is natural.
// ---------------------------------------------------------------------------
__global__ __launch_bounds__(256)
void conv_silu_kernel(const float* __restrict__ xz,
                      const float* __restrict__ cwf, const float* __restrict__ cbf,
                      const float* __restrict__ cwr, const float* __restrict__ cbr,
                      float* __restrict__ xt_f, float* __restrict__ xt_r)
{
    int idx = blockIdx.x * blockDim.x + threadIdx.x;
    if (idx >= BATCH * SEQLEN * DINNER) return;
    const int d = idx % DINNER;
    const int l = (idx / DINNER) % SEQLEN;
    const int b = idx / (DINNER * SEQLEN);

    float accf = cbf[d];
    float accr = cbr[d];
#pragma unroll
    for (int k = 0; k < 4; ++k) {
        int ls = l - 3 + k;                           // causal (left-pad 3)
        if (ls >= 0) {
            accf += cwf[d * 4 + k] * xz[((size_t)(b * SEQLEN + ls)) * (2 * DINNER) + d];
            accr += cwr[d * 4 + k] * xz[((size_t)(b * SEQLEN + (SEQLEN - 1 - ls))) * (2 * DINNER) + d];
        }
    }
    xt_f[idx] = silu_f(accf);
    xt_r[idx] = silu_f(accr);
}

// ---------------------------------------------------------------------------
// Selective scan, both directions.  One thread per (batch, channel d); 16
// states live in VGPRs; forward pass writes ysum, reverse pass adds into the
// flipped position (same thread owns both -> ordered, race-free).
// x_dbl row layout: [dt(48) | B(16) | C(16)]
// ---------------------------------------------------------------------------
__global__ __launch_bounds__(256)
void scan_kernel(const float* __restrict__ xz,
                 const float* __restrict__ xt_f, const float* __restrict__ xdbl_f,
                 const float* __restrict__ dlt_f, const float* __restrict__ Alog_f,
                 const float* __restrict__ Dp_f,
                 const float* __restrict__ xt_r, const float* __restrict__ xdbl_r,
                 const float* __restrict__ dlt_r, const float* __restrict__ Alog_r,
                 const float* __restrict__ Dp_r,
                 float* __restrict__ ysum)
{
    const int bpb = DINNER / 256;                      // blocks per batch = 6
    const int b = blockIdx.x / bpb;
    const int d = (blockIdx.x % bpb) * 256 + threadIdx.x;

    float A[DSTATE], s[DSTATE], Bm[DSTATE], Cm[DSTATE];

    // ---------------- forward direction ----------------
#pragma unroll
    for (int n = 0; n < DSTATE; ++n) { A[n] = -__expf(Alog_f[d * DSTATE + n]); s[n] = 0.f; }
    float Dd = Dp_f[d];
    for (int l = 0; l < SEQLEN; ++l) {
        const size_t rb = (size_t)(b * SEQLEN + l);
        float dlt = dlt_f[rb * DINNER + d];
        float u   = xt_f[rb * DINNER + d];
        if (l + 1 < SEQLEN) {                          // stream-ahead hints
            __builtin_prefetch(dlt_f + (rb + 1) * DINNER + d, 0, 0);
            __builtin_prefetch(xt_f  + (rb + 1) * DINNER + d, 0, 0);
        }
        const float* bc = xdbl_f + rb * (DTRANK + 2 * DSTATE);
#pragma unroll
        for (int q = 0; q < 4; ++q) {
            float4 vb = *(const float4*)(bc + DTRANK + q * 4);
            float4 vc = *(const float4*)(bc + DTRANK + DSTATE + q * 4);
            Bm[q*4+0]=vb.x; Bm[q*4+1]=vb.y; Bm[q*4+2]=vb.z; Bm[q*4+3]=vb.w;
            Cm[q*4+0]=vc.x; Cm[q*4+1]=vc.y; Cm[q*4+2]=vc.z; Cm[q*4+3]=vc.w;
        }
        float du = dlt * u, y = 0.f;
#pragma unroll
        for (int n = 0; n < DSTATE; ++n) {
            float dA = __expf(dlt * A[n]);
            s[n] = dA * s[n] + du * Bm[n];
            y += s[n] * Cm[n];
        }
        float z = xz[rb * (2 * DINNER) + DINNER + d];
        ysum[rb * DINNER + d] = (y + u * Dd) * silu_f(z);
    }

    // ---------------- reverse direction ----------------
#pragma unroll
    for (int n = 0; n < DSTATE; ++n) { A[n] = -__expf(Alog_r[d * DSTATE + n]); s[n] = 0.f; }
    Dd = Dp_r[d];
    for (int t = 0; t < SEQLEN; ++t) {                 // t = position in flipped sequence
        const size_t rb  = (size_t)(b * SEQLEN + t);               // reversed coords
        const size_t rbo = (size_t)(b * SEQLEN + (SEQLEN - 1 - t)); // original coords
        float dlt = dlt_r[rb * DINNER + d];
        float u   = xt_r[rb * DINNER + d];
        if (t + 1 < SEQLEN) {
            __builtin_prefetch(dlt_r + (rb + 1) * DINNER + d, 0, 0);
            __builtin_prefetch(xt_r  + (rb + 1) * DINNER + d, 0, 0);
        }
        const float* bc = xdbl_r + rb * (DTRANK + 2 * DSTATE);
#pragma unroll
        for (int q = 0; q < 4; ++q) {
            float4 vb = *(const float4*)(bc + DTRANK + q * 4);
            float4 vc = *(const float4*)(bc + DTRANK + DSTATE + q * 4);
            Bm[q*4+0]=vb.x; Bm[q*4+1]=vb.y; Bm[q*4+2]=vb.z; Bm[q*4+3]=vb.w;
            Cm[q*4+0]=vc.x; Cm[q*4+1]=vc.y; Cm[q*4+2]=vc.z; Cm[q*4+3]=vc.w;
        }
        float du = dlt * u, y = 0.f;
#pragma unroll
        for (int n = 0; n < DSTATE; ++n) {
            float dA = __expf(dlt * A[n]);
            s[n] = dA * s[n] + du * Bm[n];
            y += s[n] * Cm[n];
        }
        float z = xz[rbo * (2 * DINNER) + DINNER + d]; // z of the reversed stream
        ysum[rbo * DINNER + d] += (y + u * Dd) * silu_f(z);   // flip back + add
    }
}

// ---------------------------------------------------------------------------
// launch
// ---------------------------------------------------------------------------
extern "C" void kernel_launch(void* const* d_in, const int* in_sizes, int n_in,
                              void* d_out, int out_size, void* d_ws, size_t ws_size,
                              hipStream_t stream)
{
    const float* h    = (const float*)d_in[0];
    const float* Win  = (const float*)d_in[1];   // (3072, 768)
    const float* Wout = (const float*)d_in[2];   // (768, 1536)
    const float* cwf  = (const float*)d_in[3];
    const float* cbf  = (const float*)d_in[4];
    const float* xpf  = (const float*)d_in[5];   // (80, 1536)
    const float* dtwf = (const float*)d_in[6];   // (1536, 48)
    const float* dtbf = (const float*)d_in[7];
    const float* Alf  = (const float*)d_in[8];
    const float* Df   = (const float*)d_in[9];
    const float* cwr  = (const float*)d_in[10];
    const float* cbr  = (const float*)d_in[11];
    const float* xpr  = (const float*)d_in[12];
    const float* dtwr = (const float*)d_in[13];
    const float* dtbr = (const float*)d_in[14];
    const float* Alr  = (const float*)d_in[15];
    const float* Dr   = (const float*)d_in[16];

    float* ws   = (float*)d_ws;
    float* xz   = ws;                                    // 2048 x 3072
    float* xtf  = xz  + (size_t)NTOK * 2 * DINNER;       // 2048 x 1536
    float* xtr  = xtf + (size_t)NTOK * DINNER;           // 2048 x 1536
    float* xdf  = xtr + (size_t)NTOK * DINNER;           // 2048 x 80
    float* xdr  = xdf + (size_t)NTOK * (DTRANK + 2 * DSTATE);
    float* dlf  = xdr + (size_t)NTOK * (DTRANK + 2 * DSTATE);  // 2048 x 1536
    float* dlr  = dlf + (size_t)NTOK * DINNER;
    float* ysum = dlr + (size_t)NTOK * DINNER;           // 2048 x 1536
    float* out  = (float*)d_out;                         // 2048 x 768

    const dim3 wave(32);
    const int  mblk = NTOK / 16;                         // 128

    // 1) shared in_proj GEMM (reverse direction reads it flipped)
    wmma_gemm_bf16<<<dim3(mblk, (2 * DINNER) / 64), wave, 0, stream>>>(
        h, Win, nullptr, xz, NTOK, 2 * DINNER, DMODEL, DMODEL, DMODEL, 2 * DINNER, 0);

    // 2) depthwise causal conv + SiLU, both directions
    {
        int total = BATCH * SEQLEN * DINNER;
        conv_silu_kernel<<<(total + 255) / 256, 256, 0, stream>>>(
            xz, cwf, cbf, cwr, cbr, xtf, xtr);
    }

    // 3) x_proj GEMMs (N = 80 -> 2 strips of 64, last clamped)
    wmma_gemm_bf16<<<dim3(mblk, 2), wave, 0, stream>>>(
        xtf, xpf, nullptr, xdf, NTOK, 80, DINNER, DINNER, DINNER, 80, 0);
    wmma_gemm_bf16<<<dim3(mblk, 2), wave, 0, stream>>>(
        xtr, xpr, nullptr, xdr, NTOK, 80, DINNER, DINNER, DINNER, 80, 0);

    // 4) dt projection + bias + softplus (K = 48: one main step + masked tail)
    wmma_gemm_bf16<<<dim3(mblk, DINNER / 64), wave, 0, stream>>>(
        xdf, dtwf, dtbf, dlf, NTOK, DINNER, DTRANK, 80, DTRANK, DINNER, 1);
    wmma_gemm_bf16<<<dim3(mblk, DINNER / 64), wave, 0, stream>>>(
        xdr, dtwr, dtbr, dlr, NTOK, DINNER, DTRANK, 80, DTRANK, DINNER, 1);

    // 5) bidirectional selective scan + gating, accumulate y_f + flip(y_r)
    scan_kernel<<<BATCH * (DINNER / 256), 256, 0, stream>>>(
        xz, xtf, xdf, dlf, Alf, Df, xtr, xdr, dlr, Alr, Dr, ysum);

    // 6) shared out_proj GEMM straight into d_out
    wmma_gemm_bf16<<<dim3(mblk, DMODEL / 64), wave, 0, stream>>>(
        ysum, Wout, nullptr, out, NTOK, DMODEL, DINNER, DINNER, DINNER, DMODEL, 0);
}
